// SLSTM_block_9955734192295
// MI455X (gfx1250) — matrix operand order repaired and checked by reading, not compiled, not run on hardware
//
#include <hip/hip_runtime.h>
#include <hip/hip_bf16.h>
#include <math.h>

// Problem constants (B,S,H,K,L) = (8,1024,512,3,6)
#define BB   8
#define SS   1024
#define HH   512
#define KSH  3
#define LL   6
#define H2   1024   // 2H
#define H4   2048   // 4H
#define H7   3584   // 7H
#define BS   (BB*SS) // 8192 rows
#define NEGV 1e25f

#define BN   64     // N-tile per block (shared B strip)
#define BK   32     // K step (one WMMA depth)

typedef __attribute__((ext_vector_type(16))) __bf16 v16bf;
typedef __attribute__((ext_vector_type(8)))  __bf16 v8bf;
typedef __attribute__((ext_vector_type(8)))  float  v8f;

__device__ __forceinline__ __bf16 f2bf(float f) {
  union { float f; unsigned u; } v; v.f = f;
  unsigned r = v.u + 0x7FFFu + ((v.u >> 16) & 1u); // round-to-nearest-even
  unsigned short h = (unsigned short)(r >> 16);
  __bf16 o; __builtin_memcpy(&o, &h, 2); return o;
}
__device__ __forceinline__ float sigm(float x) { return 1.0f / (1.0f + __expf(-x)); }

// ---------------------------------------------------------------------------
// bf16 WMMA GEMM: C[M,N] = A[M,K](bf16,row-major,lda) * Bt[N,K](bf16, N-major)
// Optional fused epilogue: C += addend (same shape) and C += bias[n].
//
// Grid: blockIdx.x = 64-wide N strip, blockIdx.y = group of 8 consecutive
// 32-row M tiles (one per wave, 256 threads = 8 wave32).
//
// B staging: the 64x32 bf16 B tile (4KB) is loaded cooperatively into LDS
// (256 threads x 16B), double-buffered, one barrier per K step. Waves read
// B fragments via ds_load_b128; A fragments load directly from global.
// This removes the per-wave global B address chains (previous version was
// spilling address pairs in the inner loop) and cuts global B traffic 8x.
// C/D layout per CDNA5 ISA 7.12.2 (wave32); A/B k is contiguous (16*(lane/16) + s).
// ---------------------------------------------------------------------------
__global__ void gemm_bf16_wmma(const __bf16* __restrict__ A, int lda,
                               const __bf16* __restrict__ Bt, int Kdim,
                               float* __restrict__ C, int M, int N,
                               const float* __restrict__ addend,
                               const float* __restrict__ bias) {
  __shared__ __bf16 Bs[2 * BN * BK];                 // 2 x 4KB double buffer

  const int tid  = threadIdx.x;
  const int lane = tid & 31;
  const int wave = tid >> 5;
  const int nStrip = blockIdx.x;                     // shared by all waves
  const int mTile  = blockIdx.y * 8 + wave;          // 32-row tile per wave

  const int mlane = lane & 15;
  const int hi    = lane >> 4;
  const int aKlo  = hi ? 16 : 0;   // A: lanes 0-15 -> K 0..15 ; lanes 16-31 -> K 16..31
  const int bKlo  = hi ? 16 : 0;   // B: lanes 0-15 -> K 0..15 ; lanes 16-31 -> K 16..31

  // Cooperative B staging: thread -> (row, 8-elem chunk); 16B, 16B-aligned
  const int ldRow = tid >> 2;                        // 0..63
  const int ldCol = (tid & 3) * 8;                   // 0,8,16,24
  const __bf16* gB = Bt + (size_t)(nStrip * BN + ldRow) * Kdim + ldCol;
  __bf16*       sB = Bs + ldRow * BK + ldCol;

  // A fragments: direct global (per-wave rows)
  const __bf16* gA0 = A + (size_t)(mTile * 32 + mlane) * lda + aKlo;
  const __bf16* gA1 = gA0 + (size_t)16 * lda;

  // Per-lane LDS fragment base: local row = 16*t + mlane, 32B contiguous
  const __bf16* fB = Bs + mlane * BK + bKlo;

  const v8f vz = {0,0,0,0,0,0,0,0};
  v8f acc00 = vz, acc01 = vz, acc02 = vz, acc03 = vz;
  v8f acc10 = vz, acc11 = vz, acc12 = vz, acc13 = vz;

  // Preload tile 0
  *(v8bf*)sB = *(const v8bf*)gB;
  __syncthreads();

  const int nSteps = Kdim >> 5;
  for (int i = 0; i < nSteps; ++i) {
    const int k   = i << 5;
    const int cur = (i & 1) * (BN * BK);

    // Issue next B tile's global load early (overlaps with WMMAs)
    v8bf nxt;
    const bool hasNext = (i + 1) < nSteps;           // uniform
    if (hasNext) nxt = *(const v8bf*)(gB + k + BK);

    // A fragments (global)
    v8bf a0lo = *(const v8bf*)(gA0 + k);
    v8bf a0hi = *(const v8bf*)(gA0 + k + 8);
    v16bf a0 = __builtin_shufflevector(a0lo, a0hi, 0,1,2,3,4,5,6,7,8,9,10,11,12,13,14,15);
    v8bf a1lo = *(const v8bf*)(gA1 + k);
    v8bf a1hi = *(const v8bf*)(gA1 + k + 8);
    v16bf a1 = __builtin_shufflevector(a1lo, a1hi, 0,1,2,3,4,5,6,7,8,9,10,11,12,13,14,15);

    // B fragments (LDS, current buffer)
    v16bf b0 = *(const v16bf*)(fB + cur);
    v16bf b1 = *(const v16bf*)(fB + cur + 16 * BK);
    v16bf b2 = *(const v16bf*)(fB + cur + 32 * BK);
    v16bf b3 = *(const v16bf*)(fB + cur + 48 * BK);

    acc00 = __builtin_amdgcn_wmma_f32_16x16x32_bf16(false, a0, false, b0, (short)0, acc00, false, false);
    acc01 = __builtin_amdgcn_wmma_f32_16x16x32_bf16(false, a0, false, b1, (short)0, acc01, false, false);
    acc02 = __builtin_amdgcn_wmma_f32_16x16x32_bf16(false, a0, false, b2, (short)0, acc02, false, false);
    acc03 = __builtin_amdgcn_wmma_f32_16x16x32_bf16(false, a0, false, b3, (short)0, acc03, false, false);
    acc10 = __builtin_amdgcn_wmma_f32_16x16x32_bf16(false, a1, false, b0, (short)0, acc10, false, false);
    acc11 = __builtin_amdgcn_wmma_f32_16x16x32_bf16(false, a1, false, b1, (short)0, acc11, false, false);
    acc12 = __builtin_amdgcn_wmma_f32_16x16x32_bf16(false, a1, false, b2, (short)0, acc12, false, false);
    acc13 = __builtin_amdgcn_wmma_f32_16x16x32_bf16(false, a1, false, b3, (short)0, acc13, false, false);

    // Publish next tile into the other buffer; barrier retires both the
    // producer->consumer edge (next buffer) and consumer->producer reuse edge
    if (hasNext) *(v8bf*)(sB + (((i + 1) & 1) * (BN * BK))) = nxt;
    __syncthreads();
  }

  // Bias per output column (n depends only on t)
  const int n0 = nStrip * BN + mlane;
  float bv0 = 0.f, bv1 = 0.f, bv2 = 0.f, bv3 = 0.f;
  if (bias) { bv0 = bias[n0]; bv1 = bias[n0 + 16]; bv2 = bias[n0 + 32]; bv3 = bias[n0 + 48]; }

  // C/D layout: VGPR r holds M = r (lanes 0-15) or 8+r (lanes 16-31), N = lane&15
  const int mBase0 = mTile * 32 + hi * 8;
#pragma unroll
  for (int r = 0; r < 8; ++r) {
    size_t cb0 = (size_t)(mBase0 + r) * N;        // A-fragment 0 rows
    size_t cb1 = (size_t)(mBase0 + 16 + r) * N;   // A-fragment 1 rows
    float v00 = acc00[r] + bv0, v01 = acc01[r] + bv1, v02 = acc02[r] + bv2, v03 = acc03[r] + bv3;
    float v10 = acc10[r] + bv0, v11 = acc11[r] + bv1, v12 = acc12[r] + bv2, v13 = acc13[r] + bv3;
    if (addend) {
      v00 += addend[cb0 + n0];      v01 += addend[cb0 + n0 + 16];
      v02 += addend[cb0 + n0 + 32]; v03 += addend[cb0 + n0 + 48];
      v10 += addend[cb1 + n0];      v11 += addend[cb1 + n0 + 16];
      v12 += addend[cb1 + n0 + 32]; v13 += addend[cb1 + n0 + 48];
    }
    C[cb0 + n0] = v00; C[cb0 + n0 + 16] = v01; C[cb0 + n0 + 32] = v02; C[cb0 + n0 + 48] = v03;
    C[cb1 + n0] = v10; C[cb1 + n0 + 16] = v11; C[cb1 + n0 + 32] = v12; C[cb1 + n0 + 48] = v13;
  }
}

// ---------------------------------------------------------------------------
// W[K,N] fp32 -> Wt[N,K] bf16 (one-time)
// ---------------------------------------------------------------------------
__global__ void transpose_to_bf16(const float* __restrict__ W, __bf16* __restrict__ Wt,
                                  int Kdim, int Ndim) {
  int i = blockIdx.x * blockDim.x + threadIdx.x;
  if (i >= Kdim * Ndim) return;
  int n = i / Kdim, k = i - n * Kdim;
  Wt[i] = f2bf(W[(size_t)k * Ndim + n]);
}

// Initial word_h -> first half of A2 (row stride H2) for the Ux GEMM
__global__ void h_to_A2(const float* __restrict__ h, __bf16* __restrict__ A2) {
  int i = blockIdx.x * blockDim.x + threadIdx.x;   // BS*HH
  int r = i >> 9, c = i & 511;
  A2[(size_t)r * H2 + c] = f2bf(h[i]);
}

__global__ void copy_f32(const float* __restrict__ src, float* __restrict__ dst, int n) {
  int i = blockIdx.x * blockDim.x + threadIdx.x;
  if (i < n) dst[i] = src[i];
}

__global__ void seqlen_kernel(const float* __restrict__ mask, float* __restrict__ seqlen) {
  int b = threadIdx.x;
  if (b < BB) {
    float s = 0.f;
    for (int i = 0; i < SS; ++i) s += 1.0f - mask[b * SS + i];
    seqlen[b] = s;
  }
}

// out[b,h] = sum_s x[b,s,h] / seqlen[b]; optional duplicate into out2
__global__ void reduce_mean_kernel(const float* __restrict__ x, const float* __restrict__ seqlen,
                                   float* __restrict__ out, float* __restrict__ out2) {
  int b = blockIdx.x;
  float inv = 1.0f / seqlen[b];
  for (int h = threadIdx.x; h < HH; h += blockDim.x) {
    size_t base = (size_t)(b * SS) * HH + h;
    float s = 0.f;
    for (int i = 0; i < SS; ++i) s += x[base + (size_t)i * HH];
    float v = s * inv;
    out[b * HH + h] = v;
    if (out2) out2[b * HH + h] = v;
  }
}

// Build A_cat[BS, 4H] = [h | shift_before(h)*sm | shift_after(h)*sm | dgh*sm] in bf16
__global__ void acat_kernel(const float* __restrict__ wh, const float* __restrict__ mask,
                            const float* __restrict__ dgh, __bf16* __restrict__ A) {
  int idx = blockIdx.x * blockDim.x + threadIdx.x;   // r*H4 + c
  int r = idx >> 11, c = idx & 2047;
  int b = r >> 10,  s = r & 1023;
  int seg = c >> 9, h = c & 511;
  float sm = 1.0f - mask[r];
  size_t colBase = (size_t)(b * SS) * HH + h;
  float v;
  if (seg == 0) {
    v = wh[(size_t)r * HH + h];
  } else if (seg == 1) {
    v = 0.f;
#pragma unroll
    for (int k = 1; k <= KSH; ++k)
      if (s - k >= 0) v += wh[colBase + (size_t)(s - k) * HH];
    v *= sm;
  } else if (seg == 2) {
    v = 0.f;
#pragma unroll
    for (int k = 1; k <= KSH; ++k)
      if (s + k < SS) v += wh[colBase + (size_t)(s + k) * HH];
    v *= sm;
  } else {
    v = dgh[b * HH + h] * sm;
  }
  A[idx] = f2bf(v);
}

// Gate epilogue: gates (already = A@WV + WV_b + Ux) -> softmax-of-sigmoids cell
// update. Writes new word_h (fp32), new word_c (fp32, ping-pong) and the bf16
// A2 = [h_new | dgh_e] matrix for the f_ GEMM.
__global__ void gate_kernel(const float* __restrict__ gates, const float* __restrict__ cOld,
                            const float* __restrict__ mask, const float* __restrict__ dgc,
                            const float* __restrict__ dgh,
                            float* __restrict__ whOut, float* __restrict__ cNew,
                            __bf16* __restrict__ A2) {
  int idx = blockIdx.x * blockDim.x + threadIdx.x;   // r*HH + h
  int r = idx >> 9, h = idx & 511;
  int b = r >> 10, s = r & 1023;
  float sm = 1.0f - mask[r];

  size_t g0 = (size_t)r * H7 + h;
  float v[7];
#pragma unroll
  for (int j = 0; j < 7; ++j) v[j] = gates[g0 + (size_t)j * HH];
  float go = sigm(v[5]);
  float gu = tanhf(v[6]);

  float e[5], mx = -1e30f, den = 0.f;
#pragma unroll
  for (int j = 0; j < 5; ++j) { e[j] = sigm(v[j]); mx = fmaxf(mx, e[j]); }
#pragma unroll
  for (int j = 0; j < 5; ++j) { e[j] = __expf(e[j] - mx); den += e[j]; }
  float inv = 1.0f / den;

  size_t colBase = (size_t)(b * SS) * HH + h;
  float cb = 0.f, ca = 0.f;
#pragma unroll
  for (int k = 1; k <= KSH; ++k) {
    if (s - k >= 0) cb += cOld[colBase + (size_t)(s - k) * HH];
    if (s + k < SS) ca += cOld[colBase + (size_t)(s + k) * HH];
  }
  cb *= sm; ca *= sm;
  float cf   = cOld[(size_t)r * HH + h];
  float dgce = dgc[b * HH + h];

  float cn = (gu * e[0] + cb * e[1] + ca * e[2] + cf * e[3] + dgce * e[4]) * inv;
  float hn = go * tanhf(cn);
  hn *= sm; cn *= sm;

  whOut[(size_t)r * HH + h] = hn;
  cNew [(size_t)r * HH + h] = cn;
  A2[(size_t)r * H2 + h]      = f2bf(hn);
  A2[(size_t)r * H2 + HH + h] = f2bf(dgh[b * HH + h] * sm);
}

// fo = [dgh, h_avg](8x1024) @ WUfo_w(1024x1024) + b ; split -> sigmoid
__global__ void fo_kernel(const float* __restrict__ dgh, const float* __restrict__ havg,
                          const float* __restrict__ W,   const float* __restrict__ bvec,
                          float* __restrict__ dgf, float* __restrict__ dgo) {
  int b = blockIdx.x;
  const float* x0 = dgh  + b * HH;
  const float* x1 = havg + b * HH;
  for (int n = threadIdx.x; n < H2; n += blockDim.x) {
    float acc = bvec[n];
    for (int k = 0; k < HH; ++k) acc += x0[k] * W[(size_t)k * H2 + n];
    for (int k = 0; k < HH; ++k) acc += x1[k] * W[(size_t)(k + HH) * H2 + n];
    float sv = sigm(acc);
    if (n < HH) dgf[b * HH + n] = sv;
    else        dgo[b * HH + (n - HH)] = sv;
  }
}

// Column softmax over (S+1) entries per (b,h): sigmoid(fraw+bias)+mask_score
// vs dg_f, then dgc' = sum(g_f*word_c) + g_g*dgc ; dgh' = dg_o*tanh(dgc')
__global__ void colsoftmax_kernel(const float* __restrict__ fraw, const float* __restrict__ WUfb,
                                  const float* __restrict__ mask, const float* __restrict__ wc,
                                  const float* __restrict__ dgf,  const float* __restrict__ dgo,
                                  const float* __restrict__ dgcIn,
                                  float* __restrict__ dgcOut, float* __restrict__ dghOut) {
  int c = blockIdx.x * blockDim.x + threadIdx.x;   // 0..BB*HH-1
  int b = c >> 9, h = c & 511;
  float bias = WUfb[h];
  float xdg  = dgf[c];
  const float* mrow = mask + b * SS;
  size_t base = (size_t)(b * SS) * HH + h;

  float mx = xdg;
  for (int s = 0; s < SS; ++s) {
    float x = sigm(fraw[base + (size_t)s * HH] + bias) - mrow[s] * NEGV;
    mx = fmaxf(mx, x);
  }
  float gg  = __expf(xdg - mx);
  float den = gg, num = 0.f;
  for (int s = 0; s < SS; ++s) {
    float x = sigm(fraw[base + (size_t)s * HH] + bias) - mrow[s] * NEGV;
    float e = __expf(x - mx);
    den += e;
    num += e * wc[base + (size_t)s * HH];
  }
  float inv = 1.0f / den;
  float dn  = num * inv + (gg * inv) * dgcIn[c];
  dgcOut[c] = dn;
  dghOut[c] = dgo[c] * tanhf(dn);
}

// ---------------------------------------------------------------------------
extern "C" void kernel_launch(void* const* d_in, const int* in_sizes, int n_in,
                              void* d_out, int out_size, void* d_ws, size_t ws_size,
                              hipStream_t stream) {
  (void)in_sizes; (void)n_in; (void)out_size; (void)ws_size;
  const float* word_h = (const float*)d_in[0];
  const float* mask   = (const float*)d_in[1];
  /* d_in[2] = src_tokens (unused by reference math) */
  const float* WV_w   = (const float*)d_in[3];
  const float* WV_b   = (const float*)d_in[4];
  const float* U_w    = (const float*)d_in[5];
  const float* WUfo_w = (const float*)d_in[6];
  const float* WUfo_b = (const float*)d_in[7];
  const float* WUf_w  = (const float*)d_in[8];
  const float* WUf_b  = (const float*)d_in[9];

  // Output layout: word_h | word_c | dgh | dgc
  float* wh_out  = (float*)d_out;
  float* wc_out  = wh_out + (size_t)BS * HH;
  float* dgh_out = wc_out + (size_t)BS * HH;
  float* dgc_out = dgh_out + BB * HH;

  // Workspace carve-up (256B aligned)
  char* ws = (char*)d_ws;
  size_t off = 0;
  auto alloc = [&](size_t bytes) -> void* {
    void* p = ws + off; off += (bytes + 255) & ~(size_t)255; return p;
  };
  __bf16* WvT   = (__bf16*)alloc((size_t)H4 * H7 * 2);  // WV_w^T  (3584 x 2048)
  __bf16* UT    = (__bf16*)alloc((size_t)HH * H7 * 2);  // U_w^T   (3584 x 512)
  __bf16* WUfT  = (__bf16*)alloc((size_t)H2 * HH * 2);  // WUf_w^T (512 x 1024)
  float*  Ux    = (float*) alloc((size_t)BS * H7 * 4);
  float*  gates = (float*) alloc((size_t)BS * H7 * 4);
  __bf16* Acat  = (__bf16*)alloc((size_t)BS * H4 * 2);
  __bf16* A2    = (__bf16*)alloc((size_t)BS * H2 * 2);
  float*  fraw  = (float*) alloc((size_t)BS * HH * 4);
  float*  c0    = (float*) alloc((size_t)BS * HH * 4);
  float*  c1    = (float*) alloc((size_t)BS * HH * 4);
  float*  seqlen= (float*) alloc(BB * 4);
  float*  dgh   = (float*) alloc(BB * HH * 4);
  float*  dgc   = (float*) alloc(BB * HH * 4);
  float*  havg  = (float*) alloc(BB * HH * 4);
  float*  dgf   = (float*) alloc(BB * HH * 4);
  float*  dgo   = (float*) alloc(BB * HH * 4);

  const int T = 256;  // 8 waves per workgroup (wave32)
  const dim3 gridBig(H7 / BN, BS / (32 * 8));  // gates / Ux GEMMs: 56 x 32
  const dim3 gridF  (HH / BN, BS / (32 * 8));  // f_ GEMM:          8 x 32

  // One-time weight conversion (transposed, bf16)
  transpose_to_bf16<<<(H4 * H7 + T - 1) / T, T, 0, stream>>>(WV_w, WvT, H4, H7);
  transpose_to_bf16<<<(HH * H7 + T - 1) / T, T, 0, stream>>>(U_w, UT, HH, H7);
  transpose_to_bf16<<<(H2 * HH + T - 1) / T, T, 0, stream>>>(WUf_w, WUfT, H2, HH);

  // Init: A2[:, :H] = bf16(word_h); word_c = word_h; dgh = dgc = mean(word_h)
  h_to_A2<<<(BS * HH) / T, T, 0, stream>>>(word_h, A2);
  copy_f32<<<(BS * HH) / T, T, 0, stream>>>(word_h, c0, BS * HH);
  seqlen_kernel<<<1, 32, 0, stream>>>(mask, seqlen);
  reduce_mean_kernel<<<BB, 256, 0, stream>>>(word_h, seqlen, dgh, dgc);

  // Ux = word_h @ U_w   (M=8192, K=512, N=3584), no epilogue
  gemm_bf16_wmma<<<gridBig, T, 0, stream>>>(A2, H2, UT, HH, Ux, BS, H7, nullptr, nullptr);

  for (int l = 0; l < LL; ++l) {
    const float* whSrc = (l == 0) ? word_h : wh_out;
    const float* cSrc  = (l & 1) ? c1 : c0;
    float* cDst        = (l == LL - 1) ? wc_out : ((l & 1) ? c0 : c1);
    float* dgcO        = (l == LL - 1) ? dgc_out : dgc;
    float* dghO        = (l == LL - 1) ? dgh_out : dgh;

    // gates GEMM: (8192 x 2048) @ (2048 x 3584), fused += Ux + WV_b
    acat_kernel<<<(BS * H4) / T, T, 0, stream>>>(whSrc, mask, dgh, Acat);
    gemm_bf16_wmma<<<gridBig, T, 0, stream>>>(Acat, H4, WvT, H4, gates, BS, H7, Ux, WV_b);
    gate_kernel<<<(BS * HH) / T, T, 0, stream>>>(gates, cSrc, mask, dgc, dgh,
                                                 wh_out, cDst, A2);

    // global-gate path
    reduce_mean_kernel<<<BB, 256, 0, stream>>>(wh_out, seqlen, havg, nullptr);
    fo_kernel<<<BB, 256, 0, stream>>>(dgh, havg, WUfo_w, WUfo_b, dgf, dgo);

    // f_ GEMM: (8192 x 1024) @ (1024 x 512)
    gemm_bf16_wmma<<<gridF, T, 0, stream>>>(A2, H2, WUfT, H2, fraw, BS, HH, nullptr, nullptr);
    colsoftmax_kernel<<<(BB * HH) / T, T, 0, stream>>>(fraw, WUf_b, mask, cDst, dgf, dgo,
                                                       dgc, dgcO, dghO);
  }
}
